// BotRGCN_12086037971062
// MI455X (gfx1250) — matrix operand order repaired and checked by reading, not compiled, not run on hardware
//
#include <hip/hip_runtime.h>
#include <hip/hip_bf16.h>

typedef __attribute__((ext_vector_type(16))) __bf16 v16bf;
typedef __attribute__((ext_vector_type(8)))  __bf16 v8bf;
typedef __attribute__((ext_vector_type(8)))  float  v8f;

#define XDIM 1552
#define HID 128

__device__ __forceinline__ float leaky(float v) {
    return v >= 0.0f ? v : 0.01f * v;
}

__device__ __forceinline__ v8f wmma_bf16(v16bf a, v16bf b, v8f c) {
    return __builtin_amdgcn_wmma_f32_16x16x32_bf16(
        false, a, false, b, (short)0, c, false, false);
}

__device__ __forceinline__ v16bf cat8(v8bf lo, v8bf hi) {
    return __builtin_shufflevector(lo, hi, 0, 1, 2, 3, 4, 5, 6, 7,
                                   8, 9, 10, 11, 12, 13, 14, 15);
}

// ---------------------------------------------------------------------------
// Fragment loaders (layouts per CDNA5 ISA §7.12.2, wave32)
// ---------------------------------------------------------------------------

// A-fragment (16x32 bf16) from a bf16 row-major matrix: per-lane K pattern is
// two contiguous 8-element runs -> two aligned 16-byte loads.
__device__ __forceinline__ v16bf load_a_bf16(const __bf16* A, int lda, int row0,
                                             int ntot, int col0, int lane) {
    int r = row0 + (lane & 15);
    if (r >= ntot) r = ntot - 1;                       // clamp; stores are guarded
    const __bf16* ap = A + (long)r * lda + col0 + ((lane & 16) ? 8 : 0);
    v8bf lo = *(const v8bf*)ap;                        // K = base .. base+7
    v8bf hi = *(const v8bf*)(ap + 16);                 // K = base+16 .. base+23
    return cat8(lo, hi);
}

// A-fragment from an fp32 row-major matrix (x may be only 4B aligned).
__device__ __forceinline__ v16bf load_a_f32(const float* A, int lda, int row0,
                                            int ntot, int col0, int lane) {
    int r = row0 + (lane & 15);
    if (r >= ntot) r = ntot - 1;
    const float* ap = A + (long)r * lda + col0 + ((lane & 16) ? 8 : 0);
    v16bf a;
#pragma unroll
    for (int j = 0; j < 8; ++j) {
        int kk = (j < 4) ? (2 * j) : (16 + 2 * (j - 4));
        a[2 * j]     = (__bf16)ap[kk];
        a[2 * j + 1] = (__bf16)ap[kk + 1];
    }
    return a;
}

// B-fragment from prepacked fragment-major bf16 weights:
// layout [frag][lane][16 bf16] -> one 32-byte contiguous run per lane.
__device__ __forceinline__ v16bf load_b_packed(const __bf16* P, int frag, int lane) {
    const v8bf* q = (const v8bf*)(P + ((long)frag * 32 + lane) * 16);
    return cat8(q[0], q[1]);
}

// C tile (f32): lanes 0-15 -> M=v, lanes 16-31 -> M=v+8 ; N = lane&15.
__device__ __forceinline__ void store_tile_f32(float* C, int ldc, int row0, int ntot,
                                               int col0, const float* biasTile,
                                               v8f c, int lane) {
    int col = col0 + (lane & 15);
    int rb  = (lane & 16) ? 8 : 0;
    float bv = biasTile ? biasTile[lane & 15] : 0.0f;
#pragma unroll
    for (int v = 0; v < 8; ++v) {
        int r = row0 + rb + v;
        if (r < ntot)
            C[(long)r * ldc + col] = c[v] + bv;
    }
}

// C tile -> bf16 with bias + leaky (h is consumed only as a WMMA A-operand).
__device__ __forceinline__ void store_tile_bf16(__bf16* C, int ldc, int row0, int ntot,
                                                int col0, const float* biasTile,
                                                v8f c, int lane) {
    int col = col0 + (lane & 15);
    int rb  = (lane & 16) ? 8 : 0;
    float bv = biasTile[lane & 15];
#pragma unroll
    for (int v = 0; v < 8; ++v) {
        int r = row0 + rb + v;
        if (r < ntot)
            C[(long)r * ldc + col] = (__bf16)leaky(c[v] + bv);
    }
}

// Unguarded bf16 LDS tile store (16x128 wave-private tile).
__device__ __forceinline__ void store_tile_lds(__bf16* T, int col0,
                                               const float* biasTile, v8f c, int lane) {
    int col = col0 + (lane & 15);
    int rb  = (lane & 16) ? 8 : 0;
    float bv = biasTile[lane & 15];
#pragma unroll
    for (int v = 0; v < 8; ++v)
        T[(rb + v) * HID + col] = (__bf16)leaky(c[v] + bv);
}

// ---------------------------------------------------------------------------
// Kernel 0: prepack fp32 row-major Kd x Nd weights into fragment-major bf16.
// ---------------------------------------------------------------------------
__global__ void prepack_kernel(const float* __restrict__ B, int Kd, int Nd,
                               __bf16* __restrict__ out) {
    int tid = blockIdx.x * blockDim.x + threadIdx.x;
    if (tid >= Kd * Nd) return;
    int j    = tid & 15;
    int lane = (tid >> 4) & 31;
    int t    = tid >> 9;
    int NT   = Nd >> 4;
    int nt   = t % NT;
    int kt   = t / NT;
    int k    = kt * 32 + ((lane & 16) ? 8 : 0) + (j & 7) + ((j & 8) ? 16 : 0);
    int ncol = nt * 16 + (lane & 15);
    out[tid] = (__bf16)B[(long)k * Nd + ncol];
}

// ---------------------------------------------------------------------------
// Kernel 1: fused feature encoder -> h (bf16).
// ---------------------------------------------------------------------------
__global__ void encode_kernel(const float* __restrict__ x,
                              const __bf16* __restrict__ pk_des,
                              const float* __restrict__ b_des,
                              const __bf16* __restrict__ pk_tweet,
                              const float* __restrict__ b_tweet,
                              const float* __restrict__ w_num,
                              const float* __restrict__ b_num,
                              const float* __restrict__ w_cat,
                              const float* __restrict__ b_cat,
                              const __bf16* __restrict__ pk_in,
                              const float* __restrict__ b_in,
                              __bf16* __restrict__ h, int n) {
    __shared__ __bf16 hl[8][16 * HID];                 // 32 KB, wave-private tiles
    int wave = threadIdx.x >> 5;
    int lane = threadIdx.x & 31;
    int row0 = blockIdx.x * 128 + wave * 16;
    if (row0 >= n) return;                             // no block barriers here
    __bf16* tile = hl[wave];

    // des -> cols 0..31, tweet -> cols 32..63 (768->32, 24 bf16 WMMA K-steps)
#pragma unroll
    for (int half = 0; half < 2; ++half) {
        const __bf16* P  = half ? pk_tweet : pk_des;
        const float*  Bv = half ? b_tweet  : b_des;
        int xcol = half ? 5 : (5 + 768 + 11);          // tweet @5, des @784
        int hcol = half ? 32 : 0;
#pragma unroll
        for (int nt = 0; nt < 2; ++nt) {
            v8f acc = {};
            for (int kt = 0; kt < 24; ++kt) {
                v16bf a = load_a_f32(x, XDIM, row0, n, xcol + kt * 32, lane);
                v16bf b = load_b_packed(P, kt * 2 + nt, lane);
                acc = wmma_bf16(a, b, acc);
            }
            store_tile_lds(tile, hcol + nt * 16, Bv + nt * 16, acc, lane);
        }
    }

    // num (x cols 0..4) -> cols 64..95 ; cat (x cols 773..783) -> cols 96..127
    for (int rr = 0; rr < 16; ++rr) {
        int r = row0 + rr; if (r >= n) r = n - 1;
        const float* xr = x + (long)r * XDIM;
        float sn = b_num[lane];
#pragma unroll
        for (int i = 0; i < 5; ++i)  sn += xr[i] * w_num[i * 32 + lane];
        tile[rr * HID + 64 + lane] = (__bf16)leaky(sn);
        float sc = b_cat[lane];
#pragma unroll
        for (int i = 0; i < 11; ++i) sc += xr[773 + i] * w_cat[i * 32 + lane];
        tile[rr * HID + 96 + lane] = (__bf16)leaky(sc);
    }

    // Cross-lane LDS handoff within the wave.
    asm volatile("s_wait_dscnt 0" ::: "memory");

    // h_tile(16x128 bf16, LDS) @ w_in(128x128): A-frags via ds_load_b128 pairs.
    v16bf afr[4];
#pragma unroll
    for (int k = 0; k < 4; ++k)
        afr[k] = load_a_bf16(tile, HID, 0, 16, k * 32, lane);
#pragma unroll
    for (int nt = 0; nt < 8; ++nt) {
        v8f acc = {};
#pragma unroll
        for (int k = 0; k < 4; ++k)
            acc = wmma_bf16(afr[k], load_b_packed(pk_in, k * 8 + nt, lane), acc);
        store_tile_bf16(h, HID, row0, n, nt * 16, b_in + nt * 16, acc, lane);
    }
}

// ---------------------------------------------------------------------------
// Kernel 2: fused per-layer triple GEMM sharing the A stream:
//   out = A @ root + bias ; p0 = A @ W0 ; p1 = A @ W1
// A-fragments are loaded once per 16-row tile; only 3 accumulators live per
// N-tile iteration.
// ---------------------------------------------------------------------------
__global__ void gemm3_kernel(const __bf16* __restrict__ A,
                             const __bf16* __restrict__ Broot,
                             const __bf16* __restrict__ Bw0,
                             const __bf16* __restrict__ Bw1,
                             const float* __restrict__ bias,
                             float* __restrict__ out,
                             float* __restrict__ p0,
                             float* __restrict__ p1, int n) {
    int wave = threadIdx.x >> 5;
    int lane = threadIdx.x & 31;
    int row0 = blockIdx.x * 128 + wave * 16;
    if (row0 >= n) return;

    v16bf afr[4];
#pragma unroll
    for (int k = 0; k < 4; ++k)
        afr[k] = load_a_bf16(A, HID, row0, n, k * 32, lane);
#pragma unroll
    for (int nt = 0; nt < 8; ++nt) {
        v8f aR = {}, a0 = {}, a1 = {};
#pragma unroll
        for (int k = 0; k < 4; ++k) {
            int f = k * 8 + nt;
            aR = wmma_bf16(afr[k], load_b_packed(Broot, f, lane), aR);
            a0 = wmma_bf16(afr[k], load_b_packed(Bw0,   f, lane), a0);
            a1 = wmma_bf16(afr[k], load_b_packed(Bw1,   f, lane), a1);
        }
        store_tile_f32(out, HID, row0, n, nt * 16, bias + nt * 16, aR, lane);
        store_tile_f32(p0,  HID, row0, n, nt * 16, nullptr,        a0, lane);
        store_tile_f32(p1,  HID, row0, n, nt * 16, nullptr,        a1, lane);
    }
}

// ---------------------------------------------------------------------------
// Kernel 3: single-B GEMM for the classifier head -> d_out (f32).
// ---------------------------------------------------------------------------
__global__ void gemm128_kernel(const __bf16* __restrict__ A,
                               const __bf16* __restrict__ Bp,
                               const float* __restrict__ bias,
                               float* __restrict__ C, int n) {
    int wave = threadIdx.x >> 5;
    int lane = threadIdx.x & 31;
    int row0 = blockIdx.x * 128 + wave * 16;
    if (row0 >= n) return;

    v16bf afr[4];
#pragma unroll
    for (int k = 0; k < 4; ++k)
        afr[k] = load_a_bf16(A, HID, row0, n, k * 32, lane);
#pragma unroll
    for (int nt = 0; nt < 8; ++nt) {
        v8f acc = {};
#pragma unroll
        for (int k = 0; k < 4; ++k)
            acc = wmma_bf16(afr[k], load_b_packed(Bp, k * 8 + nt, lane), acc);
        store_tile_f32(C, HID, row0, n, nt * 16, bias + nt * 16, acc, lane);
    }
}

// ---------------------------------------------------------------------------
// Kernel 4: zero both relation accumulators + counts.
// ---------------------------------------------------------------------------
__global__ void zero_kernel(float* __restrict__ acc, long nacc,
                            float* __restrict__ cnt, long ncnt) {
    long i = (long)blockIdx.x * blockDim.x + threadIdx.x;
    if (i < nacc) acc[i] = 0.0f;
    if (i < ncnt) cnt[i] = 0.0f;
}

// ---------------------------------------------------------------------------
// Kernel 5: single-pass edge scatter over BOTH relations.
// One wave per edge; 4 floats/lane; f32 atomics into the L2-resident acc.
// ---------------------------------------------------------------------------
__global__ void scatter_kernel(const float* __restrict__ p0,
                               const float* __restrict__ p1,
                               const int* __restrict__ src, const int* __restrict__ dst,
                               const int* __restrict__ etype,
                               float* __restrict__ acc0, float* __restrict__ acc1,
                               float* __restrict__ cnt0, float* __restrict__ cnt1,
                               int E) {
    long gid = (long)blockIdx.x * blockDim.x + threadIdx.x;
    int e    = (int)(gid >> 5);
    int lane = (int)(gid & 31);
    if (e >= E) return;
    int rel = etype[e];
    const float* pr = rel ? p1 : p0;
    float* ar       = rel ? acc1 : acc0;
    float* cr       = rel ? cnt1 : cnt0;
    int s = src[e], d = dst[e];
    float4 v = ((const float4*)(pr + (long)s * HID))[lane];
    float* ad = ar + (long)d * HID + lane * 4;
    unsafeAtomicAdd(ad + 0, v.x);
    unsafeAtomicAdd(ad + 1, v.y);
    unsafeAtomicAdd(ad + 2, v.z);
    unsafeAtomicAdd(ad + 3, v.w);
    if (lane == 0) unsafeAtomicAdd(cr + d, 1.0f);
}

// ---------------------------------------------------------------------------
// Kernel 6: h(bf16) = leaky(out + acc0/max(c0,1) + acc1/max(c1,1)).
// ---------------------------------------------------------------------------
__global__ void combine_kernel(const float* __restrict__ out,
                               const float* __restrict__ acc0,
                               const float* __restrict__ acc1,
                               const float* __restrict__ cnt0,
                               const float* __restrict__ cnt1,
                               __bf16* __restrict__ hnew, int n) {
    long i = (long)blockIdx.x * blockDim.x + threadIdx.x;
    long tot = (long)n * HID;
    if (i >= tot) return;
    int node = (int)(i >> 7);
    float c0 = cnt0[node]; if (c0 < 1.0f) c0 = 1.0f;
    float c1 = cnt1[node]; if (c1 < 1.0f) c1 = 1.0f;
    float v = out[i] + acc0[i] / c0 + acc1[i] / c1;
    hnew[i] = (__bf16)leaky(v);
}

extern "C" void kernel_launch(void* const* d_in, const int* in_sizes, int n_in,
                              void* d_out, int out_size, void* d_ws, size_t ws_size,
                              hipStream_t stream) {
    const float* x       = (const float*)d_in[0];
    const int*   ei      = (const int*)  d_in[1];
    const int*   et      = (const int*)  d_in[2];
    const float* w_des   = (const float*)d_in[3];
    const float* b_des   = (const float*)d_in[4];
    const float* w_tweet = (const float*)d_in[5];
    const float* b_tweet = (const float*)d_in[6];
    const float* w_num   = (const float*)d_in[7];
    const float* b_num   = (const float*)d_in[8];
    const float* w_cat   = (const float*)d_in[9];
    const float* b_cat   = (const float*)d_in[10];
    const float* w_in    = (const float*)d_in[11];
    const float* b_in    = (const float*)d_in[12];
    const float* rg1_w   = (const float*)d_in[13];
    const float* rg1_root= (const float*)d_in[14];
    const float* rg1_b   = (const float*)d_in[15];
    const float* rg2_w   = (const float*)d_in[16];
    const float* rg2_root= (const float*)d_in[17];
    const float* rg2_b   = (const float*)d_in[18];
    const float* w_cls   = (const float*)d_in[19];
    const float* b_cls   = (const float*)d_in[20];

    const int  n = in_sizes[0] / XDIM;
    const int  E = in_sizes[1] / 2;
    const int* src = ei;
    const int* dst = ei + E;
    const long big = (long)n * HID;

    // ---- workspace layout ----
    // h (bf16) | packed weights (bf16) | out,p0,p1,acc0,acc1 (f32) | cnt0,cnt1
    __bf16* h       = (__bf16*)d_ws;
    __bf16* pk_des  = h + big;                 // 768*32  = 24576 elems
    __bf16* pk_tw   = pk_des + 24576;
    __bf16* pk_in   = pk_tw  + 24576;          // 128*128 = 16384 elems each below
    __bf16* pk_r1r  = pk_in   + 16384;
    __bf16* pk_r1w0 = pk_r1r  + 16384;
    __bf16* pk_r1w1 = pk_r1w0 + 16384;
    __bf16* pk_r2r  = pk_r1w1 + 16384;
    __bf16* pk_r2w0 = pk_r2r  + 16384;
    __bf16* pk_r2w1 = pk_r2w0 + 16384;
    __bf16* pk_cls  = pk_r2w1 + 16384;
    float*  out     = (float*)(pk_cls + 16384);
    float*  p0      = out  + big;
    float*  p1      = p0   + big;
    float*  acc0    = p1   + big;
    float*  acc1    = acc0 + big;
    float*  cnt0    = acc1 + big;
    float*  cnt1    = cnt0 + n;

    // ---- prepack weights to fragment-major bf16 (tiny, stays in L2) ----
    prepack_kernel<<<(768 * 32 + 255) / 256, 256, 0, stream>>>(w_des,   768, 32, pk_des);
    prepack_kernel<<<(768 * 32 + 255) / 256, 256, 0, stream>>>(w_tweet, 768, 32, pk_tw);
    const int g128 = (HID * HID + 255) / 256;
    prepack_kernel<<<g128, 256, 0, stream>>>(w_in,              HID, HID, pk_in);
    prepack_kernel<<<g128, 256, 0, stream>>>(rg1_root,          HID, HID, pk_r1r);
    prepack_kernel<<<g128, 256, 0, stream>>>(rg1_w,             HID, HID, pk_r1w0);
    prepack_kernel<<<g128, 256, 0, stream>>>(rg1_w + HID * HID, HID, HID, pk_r1w1);
    prepack_kernel<<<g128, 256, 0, stream>>>(rg2_root,          HID, HID, pk_r2r);
    prepack_kernel<<<g128, 256, 0, stream>>>(rg2_w,             HID, HID, pk_r2w0);
    prepack_kernel<<<g128, 256, 0, stream>>>(rg2_w + HID * HID, HID, HID, pk_r2w1);
    prepack_kernel<<<g128, 256, 0, stream>>>(w_cls,             HID, HID, pk_cls);

    const int gridM = (n + 127) / 128;
    const int gz    = (int)((2 * big + 255) / 256);
    const int gc    = (int)((big + 255) / 256);
    const int gs    = (int)(((long)E * 32 + 255) / 256);

    encode_kernel<<<gridM, 256, 0, stream>>>(x, pk_des, b_des, pk_tw, b_tweet,
                                             w_num, b_num, w_cat, b_cat,
                                             pk_in, b_in, h, n);

    for (int layer = 0; layer < 2; ++layer) {
        const __bf16* rootp = layer ? pk_r2r  : pk_r1r;
        const __bf16* w0p   = layer ? pk_r2w0 : pk_r1w0;
        const __bf16* w1p   = layer ? pk_r2w1 : pk_r1w1;
        const float*  bb    = layer ? rg2_b   : rg1_b;

        gemm3_kernel<<<gridM, 256, 0, stream>>>(h, rootp, w0p, w1p, bb,
                                                out, p0, p1, n);
        zero_kernel<<<gz, 256, 0, stream>>>(acc0, 2 * big, cnt0, 2L * n);
        scatter_kernel<<<gs, 256, 0, stream>>>(p0, p1, src, dst, et,
                                               acc0, acc1, cnt0, cnt1, E);
        combine_kernel<<<gc, 256, 0, stream>>>(out, acc0, acc1, cnt0, cnt1, h, n);
    }

    gemm128_kernel<<<gridM, 256, 0, stream>>>(h, pk_cls, b_cls, (float*)d_out, n);
}
